// SpikingMLP_45397804319541
// MI455X (gfx1250) — matrix operand order repaired and checked by reading, not compiled, not run on hardware
//
#include <hip/hip_runtime.h>

typedef __attribute__((ext_vector_type(16))) _Float16 v16h;
typedef __attribute__((ext_vector_type(8)))  _Float16 v8h;
typedef __attribute__((ext_vector_type(8)))  float    v8f;

#define B_ 8
#define C_ 256
#define T_ 4
#define N_ 4096
#define H_ 1024
#define L_ (T_ * N_)      /* 16384  */
#define J_ (B_ * L_)      /* 131072 */

#define INV_TAU (1.0f / 1.5f)
#define V_TH    1.0f
#define BN_EPS  1e-5f

// ---- workspace layout (bytes) ----
#define OFF_W1H    ((size_t)0)                       // H*C fp16 = 512 KiB
#define OFF_W2H    ((size_t)(H_ * C_ * 2))           // C*H fp16 = 512 KiB
#define OFF_SCALE1 (OFF_W2H + (size_t)C_ * H_ * 2)   // H f32
#define OFF_BIAS1  (OFF_SCALE1 + (size_t)H_ * 4)
#define OFF_SCALE2 (OFF_BIAS1 + (size_t)H_ * 4)      // C f32
#define OFF_BIAS2  (OFF_SCALE2 + (size_t)C_ * 4)
#define OFF_S1     ((size_t)2 * 1024 * 1024)                 // [J][C] fp16 = 64 MiB
#define OFF_S2     (OFF_S1 + (size_t)J_ * C_ * 2)            // [J][H] fp16 = 256 MiB

static __device__ __forceinline__ v16h cat16(v8h lo, v8h hi) {
    return __builtin_shufflevector(lo, hi, 0, 1, 2, 3, 4, 5, 6, 7,
                                           8, 9, 10, 11, 12, 13, 14, 15);
}

// ---------------------------------------------------------------------------
// Prep: fp16 weights + folded BN scale/shift
// ---------------------------------------------------------------------------
__global__ __launch_bounds__(256) void k_prep_w(const float* __restrict__ w1f,
                                                const float* __restrict__ w2f,
                                                _Float16* __restrict__ w1h,
                                                _Float16* __restrict__ w2h) {
    int i = blockIdx.x * 256 + threadIdx.x;
    if (i < H_ * C_) {
        w1h[i] = (_Float16)w1f[i];
        w2h[i] = (_Float16)w2f[i];
    }
}

__global__ __launch_bounds__(256) void k_prep_p(
    const float* __restrict__ fc1_b,
    const float* __restrict__ g1, const float* __restrict__ be1,
    const float* __restrict__ mu1, const float* __restrict__ va1,
    const float* __restrict__ g2, const float* __restrict__ be2,
    const float* __restrict__ mu2, const float* __restrict__ va2,
    float* __restrict__ scale1, float* __restrict__ bias1,
    float* __restrict__ scale2, float* __restrict__ bias2) {
    int i = blockIdx.x * 256 + threadIdx.x;
    if (i < H_) {
        float inv = g1[i] * rsqrtf(va1[i] + BN_EPS);
        scale1[i] = inv;
        bias1[i]  = inv * fc1_b[i] + (be1[i] - mu1[i] * inv);  // bn1(gemm + fc1_b)
    }
    if (i < C_) {
        float inv = g2[i] * rsqrtf(va2[i] + BN_EPS);
        scale2[i] = inv;
        bias2[i]  = be2[i] - mu2[i] * inv;
    }
}

// ---------------------------------------------------------------------------
// K1: bn2 + LIF over T, transpose [B,C,T,N] -> s1[J][C] (fp16, K-contiguous)
// Tile: one b, 64 channels, 32 n, all 4 t. LDS transpose, 128-bit writes.
// c-dim padded 64 -> 72 halves (144 B: 36-dword stride, 16-B aligned rows)
// ---------------------------------------------------------------------------
__global__ __launch_bounds__(256) void k_bn_lif1(const float* __restrict__ x,
                                                 const float* __restrict__ scale2,
                                                 const float* __restrict__ bias2,
                                                 _Float16* __restrict__ s1) {
    __shared__ __align__(16) _Float16 lds[T_][32][72];
    const int tid = threadIdx.x;
    const int n0 = blockIdx.x * 32;
    const int c0 = blockIdx.y * 64;
    const int b  = blockIdx.z;

    // load coalesced over n, run LIF serially over t
    for (int i = 0; i < 8; ++i) {
        int e = i * 256 + tid;       // 2048 (c,n) pairs
        int n = e & 31;
        int c = e >> 5;              // 0..63
        float sc = scale2[c0 + c];
        float bs = bias2[c0 + c];
        const float* xp = x + ((size_t)(b * C_ + c0 + c) * T_) * N_ + n0 + n;
        float v = 0.f;
#pragma unroll
        for (int t = 0; t < T_; ++t) {
            float xt = sc * xp[(size_t)t * N_] + bs;
            v += (xt - v) * INV_TAU;
            if (v - V_TH >= 0.f) {
                lds[t][n][c] = (_Float16)1.0f;
                v = 0.f;                        // hard reset
            } else {
                lds[t][n][c] = (_Float16)0.0f;
            }
        }
    }
    __syncthreads();
    // write: 4t*32n*64c halves = 1024 float4, 4 per thread, c-contiguous
    for (int i = 0; i < 4; ++i) {
        int idx = i * 256 + tid;     // float4 id
        int cc  = (idx & 7) * 8;     // 8 chunks of 8 halves per (t,n) row
        int row = idx >> 3;          // 0..127
        int n = row & 31;
        int t = row >> 5;
        size_t j = (size_t)b * L_ + (size_t)t * N_ + n0 + n;
        *(float4*)(s1 + j * C_ + c0 + cc) = *(const float4*)(&lds[t][n][cc]);
    }
}

// ---------------------------------------------------------------------------
// K2: fc1 (WMMA f16, K=C=256) + bn1 + LIF over T (in registers) -> s2[J][H]
// Block: (b, 16 n's x 4 t's rows, 128 h cols). 8 waves, each one h-16 strip.
// A tiles (all 4 t) staged once in LDS (32 KB); LDS reused to transpose the
// spike output so s2 is written with 128-bit coalesced stores.
// ---------------------------------------------------------------------------
__global__ __launch_bounds__(256) void k_fc1_lif2(const _Float16* __restrict__ s1,
                                                  const _Float16* __restrict__ w1,
                                                  const float* __restrict__ scale1,
                                                  const float* __restrict__ bias1,
                                                  _Float16* __restrict__ s2) {
    __shared__ __align__(16) _Float16 ldsA[T_ * 16 * C_];  // 16384 halves = 32 KB
    const int tid  = threadIdx.x;
    const int lane = tid & 31;
    const int wave = tid >> 5;
    const int n0 = blockIdx.x * 16;
    const int b  = blockIdx.y;
    const int h0 = blockIdx.z * 128;

    // stage A: rows (t, m) of s1, full K=256
#pragma unroll
    for (int i = 0; i < 8; ++i) {
        int off = (i * 256 + tid) * 8;  // half offset, 16B chunks
        int row = off >> 8;             // 0..63 -> (t, m)
        int col = off & 255;
        int t = row >> 4, m = row & 15;
        const _Float16* gp =
            s1 + ((size_t)b * L_ + (size_t)t * N_ + n0 + m) * C_ + col;
        *(float4*)(&ldsA[off]) = *(const float4*)gp;
    }
    __syncthreads();

    v8f acc[T_] = {};
    const int hcol = h0 + wave * 16 + (lane & 15);  // B-frag col (n = lane&15)
    const int kb   = (lane >> 4) * 8;               // B-frag k base within chunk
    const int ab   = (lane >> 4) * 8;               // A-frag k base within chunk
    const int m    = lane & 15;                     // A-frag row

#pragma unroll
    for (int kc = 0; kc < C_ / 32; ++kc) {
        const int c0k = kc * 32;
        const _Float16* bp = w1 + (size_t)hcol * C_ + c0k + kb;
        v16h bf = cat16(*(const v8h*)bp, *(const v8h*)(bp + 16));
        // gather all four t-fragments first (one ds clause), then 4 WMMAs
        v16h af[T_];
#pragma unroll
        for (int t = 0; t < T_; ++t) {
            const _Float16* ap = &ldsA[(t * 16 + m) * C_ + c0k + ab];
            af[t] = cat16(*(const v8h*)ap, *(const v8h*)(ap + 16));
        }
#pragma unroll
        for (int t = 0; t < T_; ++t) {
            acc[t] = __builtin_amdgcn_wmma_f32_16x16x32_f16(
                false, af[t], false, bf, (short)0, acc[t], false, false);
        }
    }

    // epilogue: bn1 + LIF across the 4 t-fragments, element-wise in registers;
    // stage spikes in (now dead) ldsA, then 128-bit coalesced stores to s2.
    const float sc = scale1[hcol];
    const float bs = bias1[hcol];
    const int mb = (lane >> 4) * 8;  // D rows: VGPR r -> M = r + 8*(lane>=16)
    const int hl = wave * 16 + (lane & 15);
    __syncthreads();                 // all waves done reading A tiles
#pragma unroll
    for (int r = 0; r < 8; ++r) {
        int mm = mb + r;
        float v = 0.f;
#pragma unroll
        for (int t = 0; t < T_; ++t) {
            float y = sc * acc[t][r] + bs;
            v += (y - v) * INV_TAU;
            _Float16 s;
            if (v - V_TH >= 0.f) { s = (_Float16)1.0f; v = 0.f; }
            else                 { s = (_Float16)0.0f; }
            ldsA[(t * 16 + mm) * 128 + hl] = s;   // [4][16][128] spike tile
        }
    }
    __syncthreads();
    // 8192 halves = 1024 float4, 4 per thread; 256B contiguous per row
#pragma unroll
    for (int i = 0; i < 4; ++i) {
        int idx  = i * 256 + tid;       // float4 id
        int hoff = (idx & 15) * 8;      // 16 chunks per 128-h row
        int row  = idx >> 4;            // 0..63 -> (t, m)
        int t = row >> 4, mm = row & 15;
        _Float16* gp =
            s2 + ((size_t)b * L_ + (size_t)t * N_ + n0 + mm) * H_ + h0 + hoff;
        *(float4*)gp = *(const float4*)(&ldsA[row * 128 + hoff]);
    }
}

// ---------------------------------------------------------------------------
// K3: fc2 (WMMA f16, K=H=1024) + bias -> out [B,C,T,N] f32
// Block: 16 j rows (staged once in LDS, 32 KB), 128 c cols (8 waves).
// ---------------------------------------------------------------------------
__global__ __launch_bounds__(256) void k_fc2(const _Float16* __restrict__ s2,
                                             const _Float16* __restrict__ w2,
                                             const float* __restrict__ fc2_b,
                                             float* __restrict__ out) {
    __shared__ __align__(16) _Float16 ldsA[16 * H_];  // 16384 halves = 32 KB
    const int tid  = threadIdx.x;
    const int lane = tid & 31;
    const int wave = tid >> 5;
    const size_t j0 = (size_t)blockIdx.x * 16;
    const int b  = (int)(j0 / L_);
    const int l0 = (int)(j0 % L_);

#pragma unroll
    for (int i = 0; i < 8; ++i) {
        int off = (i * 256 + tid) * 8;
        int m = off >> 10, col = off & 1023;
        *(float4*)(&ldsA[off]) = *(const float4*)(s2 + (j0 + m) * H_ + col);
    }
    __syncthreads();

    const int c  = blockIdx.y * 128 + wave * 16 + (lane & 15);
    const int kb = (lane >> 4) * 8;
    const int ab = (lane >> 4) * 8;
    const int m  = lane & 15;
    v8f acc = {};

#pragma unroll
    for (int kc = 0; kc < H_ / 32; ++kc) {
        const int h0k = kc * 32;
        const _Float16* bp = w2 + (size_t)c * H_ + h0k + kb;
        v16h bf = cat16(*(const v8h*)bp, *(const v8h*)(bp + 16));
        const _Float16* ap = &ldsA[m * H_ + h0k + ab];
        v16h af = cat16(*(const v8h*)ap, *(const v8h*)(ap + 16));
        acc = __builtin_amdgcn_wmma_f32_16x16x32_f16(
            false, af, false, bf, (short)0, acc, false, false);
    }

    const float bias = fc2_b[c];
    const int mb = (lane >> 4) * 8;
    float* op = out + ((size_t)b * C_ + c) * L_ + l0 + mb;  // 16B aligned
    float4 f0 = make_float4(acc[0] + bias, acc[1] + bias, acc[2] + bias, acc[3] + bias);
    float4 f1 = make_float4(acc[4] + bias, acc[5] + bias, acc[6] + bias, acc[7] + bias);
    *(float4*)(op)     = f0;
    *(float4*)(op + 4) = f1;
}

// ---------------------------------------------------------------------------
extern "C" void kernel_launch(void* const* d_in, const int* in_sizes, int n_in,
                              void* d_out, int out_size, void* d_ws, size_t ws_size,
                              hipStream_t stream) {
    (void)in_sizes; (void)n_in; (void)out_size; (void)ws_size;
    const float* x      = (const float*)d_in[0];
    const float* fc1_w  = (const float*)d_in[1];
    const float* fc1_b  = (const float*)d_in[2];
    const float* fc2_w  = (const float*)d_in[3];
    const float* fc2_b  = (const float*)d_in[4];
    const float* g1  = (const float*)d_in[5];
    const float* be1 = (const float*)d_in[6];
    const float* mu1 = (const float*)d_in[7];
    const float* va1 = (const float*)d_in[8];
    const float* g2  = (const float*)d_in[9];
    const float* be2 = (const float*)d_in[10];
    const float* mu2 = (const float*)d_in[11];
    const float* va2 = (const float*)d_in[12];
    float* out = (float*)d_out;

    char* ws = (char*)d_ws;
    _Float16* w1h    = (_Float16*)(ws + OFF_W1H);
    _Float16* w2h    = (_Float16*)(ws + OFF_W2H);
    float*    scale1 = (float*)(ws + OFF_SCALE1);
    float*    bias1  = (float*)(ws + OFF_BIAS1);
    float*    scale2 = (float*)(ws + OFF_SCALE2);
    float*    bias2  = (float*)(ws + OFF_BIAS2);
    _Float16* s1     = (_Float16*)(ws + OFF_S1);
    _Float16* s2     = (_Float16*)(ws + OFF_S2);

    k_prep_w<<<(H_ * C_ + 255) / 256, 256, 0, stream>>>(fc1_w, fc2_w, w1h, w2h);
    k_prep_p<<<(H_ + 255) / 256, 256, 0, stream>>>(
        fc1_b, g1, be1, mu1, va1, g2, be2, mu2, va2,
        scale1, bias1, scale2, bias2);

    k_bn_lif1<<<dim3(N_ / 32, C_ / 64, B_), 256, 0, stream>>>(x, scale2, bias2, s1);

    k_fc1_lif2<<<dim3(N_ / 16, B_, H_ / 128), 256, 0, stream>>>(
        s1, w1h, scale1, bias1, s2);

    k_fc2<<<dim3(J_ / 16, C_ / 128), 256, 0, stream>>>(s2, w2h, fc2_b, out);
}
